// GraphEncoder_33414845563579
// MI455X (gfx1250) — matrix operand order repaired and checked by reading, not compiled, not run on hardware
//
#include <hip/hip_runtime.h>
#include <hip/hip_bf16.h>

// ---------------------------------------------------------------------------
// Graph transformer encoder for MI455X (gfx1250, wave32, WMMA).
// N=4096, E=131072, H=256, HEADS=4, HD=64, IN_DIM=131, OUT=512, L=3.
//  - all dense matmuls via v_wmma_f32_16x16x32_f16 (f16 in, f32 acc)
//  - scores materialized per-head (64MB) in ws; adj (64MB) stays L2-hot
//  - softmax fused into AV GEMM: rowstats pass + exp-on-load + rowsum divide
//  - AV A-tiles staged to LDS with global_load_async_to_lds_b128 (dbl-buffer)
// ---------------------------------------------------------------------------

typedef __attribute__((ext_vector_type(16))) _Float16 v16h;
typedef __attribute__((ext_vector_type(8)))  float    v8f;

#define NN     4096
#define HH     256
#define EE     131072
#define INDIM  131

__device__ __forceinline__ v8f wmma32(v16h a, v16h b, v8f c) {
    return __builtin_amdgcn_wmma_f32_16x16x32_f16(
        false, a, false, b, (short)0, c, false, false);
}

__device__ __forceinline__ float gelu_exact(float v) {
    return 0.5f * v * (1.0f + erff(v * 0.70710678118654752f));
}

// A tile 16x32 (MxK), row-major source, f32 -> f16.
// lane<16: row M=lane, elems 0..7 = K[k0..k0+7], 8..15 = K[k0+16..k0+23]
// lane>=16: same row, elems 0..7 = K[k0+8..], 8..15 = K[k0+24..]
__device__ __forceinline__ v16h load_A16(const float* A, int lda, int m0, int k0, int lane) {
    int r = lane & 15, half = lane >> 4;
    const float* p = A + (size_t)(m0 + r) * lda + k0;
    v16h a;
#pragma unroll
    for (int j = 0; j < 8; ++j) {
        a[j]     = (_Float16)p[half * 8 + j];
        a[8 + j] = (_Float16)p[16 + half * 8 + j];
    }
    return a;
}

__device__ __forceinline__ v16h load_A16g(const float* A, int lda, int m0, int k0, int K, int lane) {
    int r = lane & 15, half = lane >> 4;
    const float* p = A + (size_t)(m0 + r) * lda;
    v16h a;
#pragma unroll
    for (int j = 0; j < 8; ++j) {
        int k0j = k0 + half * 8 + j;
        int k1j = k0 + 16 + half * 8 + j;
        a[j]     = (_Float16)((k0j < K) ? p[k0j] : 0.0f);
        a[8 + j] = (_Float16)((k1j < K) ? p[k1j] : 0.0f);
    }
    return a;
}

// B tile 32x16 (KxN) from row-major W[K][ldw]:
// lane<16: col N=lane, elems 0..7 = W[k0+j], 8..15 = W[k0+16+(j-8)]; lane>=16: same with +8
__device__ __forceinline__ v16h load_B16(const float* W, int ldw, int k0, int n0, int lane) {
    int c = lane & 15, half = lane >> 4;
    v16h b;
#pragma unroll
    for (int j = 0; j < 16; ++j)
        b[j] = (_Float16)W[(size_t)(k0 + half * 8 + j + (j & 8)) * ldw + n0 + c];
    return b;
}

__device__ __forceinline__ v16h load_B16g(const float* W, int ldw, int k0, int n0, int K, int lane) {
    int c = lane & 15, half = lane >> 4;
    v16h b;
#pragma unroll
    for (int j = 0; j < 16; ++j) {
        int k = k0 + half * 8 + j + (j & 8);
        b[j] = (_Float16)((k < K) ? W[(size_t)k * ldw + n0 + c] : 0.0f);
    }
    return b;
}

// B tile = row-major matrix transposed: B[k][m] = M[m0+col][k0+k] (contiguous in k)
__device__ __forceinline__ v16h load_BT16(const float* M, int ldm, int m0, int k0, int lane) {
    int c = lane & 15, half = lane >> 4;
    const float* p = M + (size_t)(m0 + c) * ldm + k0 + half * 8;
    v16h b;
#pragma unroll
    for (int j = 0; j < 16; ++j) b[j] = (_Float16)p[j + (j & 8)];
    return b;
}

// ---------------------------------------------------------------------------
// Generic fused GEMM: out[m][0..255] for a 16-row strip per block.
// MODE 0: out = A@W + bias
// MODE 1: out = gelu(LN(A@W + bias))          (encoder stage 1)
// MODE 2: out = LN(A@W + bias + resid)        (attention output projection)
// Block: 256 threads = 8 waves; wave w covers columns [32w, 32w+32).
// ---------------------------------------------------------------------------
template <int MODE>
__global__ __launch_bounds__(256) void gemm16(
    const float* __restrict__ A, int lda, int K,
    const float* __restrict__ W,
    const float* __restrict__ bias,
    const float* __restrict__ gamma, const float* __restrict__ beta,
    const float* __restrict__ resid,
    float* __restrict__ out)
{
    __shared__ float tile[16][257];
    __shared__ float red1[16][16], red2[16][16];
    __shared__ float mu[16], rstd[16];

    int m0 = blockIdx.x * 16;
    int wave = threadIdx.x >> 5, lane = threadIdx.x & 31;
    int n0 = wave * 32;

    v8f c0 = {}, c1 = {};
    int ksteps = (K + 31) >> 5;
    for (int s = 0; s < ksteps; ++s) {
        int k0 = s * 32;
        v16h a  = load_A16g(A, lda, m0, k0, K, lane);
        v16h b0 = load_B16g(W, HH, k0, n0,      K, lane);
        v16h b1 = load_B16g(W, HH, k0, n0 + 16, K, lane);
        c0 = wmma32(a, b0, c0);
        c1 = wmma32(a, b1, c1);
    }

    int col = n0 + (lane & 15);
    int rloc = (lane >> 4) * 8;
#pragma unroll
    for (int r = 0; r < 8; ++r) {
        float v0 = c0[r] + bias[col];
        float v1 = c1[r] + bias[col + 16];
        if (MODE == 2) {
            v0 += resid[(size_t)(m0 + rloc + r) * HH + col];
            v1 += resid[(size_t)(m0 + rloc + r) * HH + col + 16];
        }
        tile[rloc + r][col]      = v0;
        tile[rloc + r][col + 16] = v1;
    }
    __syncthreads();

    int t = threadIdx.x;
    if (MODE == 0) {
#pragma unroll 4
        for (int r = 0; r < 16; ++r)
            out[(size_t)(m0 + r) * HH + t] = tile[r][t];
        return;
    }

    // LayerNorm over each of the 16 rows (256 cols)
    int row = t >> 4, seg = t & 15;
    float s1 = 0.f, s2 = 0.f;
#pragma unroll
    for (int i = 0; i < 16; ++i) {
        float v = tile[row][seg + 16 * i];
        s1 += v; s2 += v * v;
    }
    red1[row][seg] = s1; red2[row][seg] = s2;
    __syncthreads();
    if (t < 16) {
        float a1 = 0.f, a2 = 0.f;
#pragma unroll
        for (int i = 0; i < 16; ++i) { a1 += red1[t][i]; a2 += red2[t][i]; }
        float m = a1 * (1.0f / 256.0f);
        float var = a2 * (1.0f / 256.0f) - m * m;
        mu[t] = m;
        rstd[t] = rsqrtf(var + 1e-5f);
    }
    __syncthreads();
    float g = gamma[t], be_ = beta[t];
#pragma unroll 4
    for (int r = 0; r < 16; ++r) {
        float v = (tile[r][t] - mu[r]) * rstd[r] * g + be_;
        if (MODE == 1) v = gelu_exact(v);
        out[(size_t)(m0 + r) * HH + t] = v;
    }
}

// ---------------------------------------------------------------------------
// S[n][m] = (Q_h[n,:] . K_h[m,:]) / 8 + 10*adj[n][m]
// grid.x = 256 n-tiles * 8 column blocks; wave handles 4 m-tiles (A reuse).
// ---------------------------------------------------------------------------
__global__ __launch_bounds__(256) void qk_kernel(
    const float* __restrict__ Q, const float* __restrict__ Km,
    const float* __restrict__ adj, float* __restrict__ S, int hoff)
{
    int ntile = blockIdx.x >> 3;
    int cb    = blockIdx.x & 7;
    int wave  = threadIdx.x >> 5, lane = threadIdx.x & 31;
    int n0 = ntile * 16;
    int m0base = cb * 512 + wave * 64;

    v16h a0 = load_A16(Q, HH, n0, hoff,      lane);
    v16h a1 = load_A16(Q, HH, n0, hoff + 32, lane);

    int rloc = (lane >> 4) * 8;
    int cl = lane & 15;
#pragma unroll
    for (int mt = 0; mt < 4; ++mt) {
        int m0 = m0base + mt * 16;
        v16h b0 = load_BT16(Km, HH, m0, hoff,      lane);
        v16h b1 = load_BT16(Km, HH, m0, hoff + 32, lane);
        v8f c = {};
        c = wmma32(a0, b0, c);
        c = wmma32(a1, b1, c);
        int col = m0 + cl;
#pragma unroll
        for (int r = 0; r < 8; ++r) {
            size_t idx = (size_t)(n0 + rloc + r) * NN + col;
            S[idx] = c[r] * 0.125f + 10.0f * adj[idx];
        }
    }
}

// edge_attn[e][h] = edge_features[e][:] . We[:,h] + be[h]
__global__ __launch_bounds__(256) void edge_attn_kernel(
    const float* __restrict__ ef, const float* __restrict__ We,
    const float* __restrict__ be, float* __restrict__ ea)
{
    int e = blockIdx.x * 256 + threadIdx.x;
    if (e >= EE) return;
    const float* f = ef + (size_t)e * 64;
    float acc0 = be[0], acc1 = be[1], acc2 = be[2], acc3 = be[3];
#pragma unroll 8
    for (int k = 0; k < 64; ++k) {
        float v = f[k];
        acc0 += v * We[k * 4 + 0];
        acc1 += v * We[k * 4 + 1];
        acc2 += v * We[k * 4 + 2];
        acc3 += v * We[k * 4 + 3];
    }
    ea[e * 4 + 0] = acc0; ea[e * 4 + 1] = acc1;
    ea[e * 4 + 2] = acc2; ea[e * 4 + 3] = acc3;
}

__global__ __launch_bounds__(256) void scatter_kernel(
    float* __restrict__ S, const int* __restrict__ src,
    const int* __restrict__ tgt, const float* __restrict__ ea, int h)
{
    int e = blockIdx.x * 256 + threadIdx.x;
    if (e >= EE) return;
    atomicAdd(&S[(size_t)src[e] * NN + tgt[e]], ea[e * 4 + h]);
}

// Per-row max and sum(exp(v-max)) over 4096 columns. One block per row.
__global__ __launch_bounds__(256) void rowstats_kernel(
    const float* __restrict__ S, float* __restrict__ rowmax, float* __restrict__ rowsum)
{
    __shared__ float red[256];
    const float* p = S + (size_t)blockIdx.x * NN;
    int t = threadIdx.x;

    float mx = -1e30f;
    for (int i = t; i < NN; i += 256) mx = fmaxf(mx, p[i]);
    red[t] = mx; __syncthreads();
    for (int s = 128; s > 0; s >>= 1) { if (t < s) red[t] = fmaxf(red[t], red[t + s]); __syncthreads(); }
    mx = red[0]; __syncthreads();

    float sum = 0.f;
    for (int i = t; i < NN; i += 256) sum += __expf(p[i] - mx);
    red[t] = sum; __syncthreads();
    for (int s = 128; s > 0; s >>= 1) { if (t < s) red[t] += red[t + s]; __syncthreads(); }
    if (t == 0) { rowmax[blockIdx.x] = mx; rowsum[blockIdx.x] = red[0]; }
}

// ---------------------------------------------------------------------------
// out[n][hoff+d] = sum_m exp(S[n][m]-rowmax[n]) * V[m][hoff+d] / rowsum[n]
// One block per 16-row tile; 8 waves split K=4096; each wave's 16x32 S tile
// is staged to LDS with global_load_async_to_lds_b128 (double-buffered,
// wave-private -> only s_wait_asynccnt needed, no barriers); exp applied when
// building the f16 A fragment; LDS reduction of wave partials at the end.
// ---------------------------------------------------------------------------
__global__ __launch_bounds__(256) void av_kernel(
    const float* __restrict__ S, const float* __restrict__ V,
    const float* __restrict__ rowmax, const float* __restrict__ rowsum,
    float* __restrict__ out, int hoff)
{
    __shared__ float astage[8][2][16 * 32];   // 32 KB: per-wave dbl-buffered S tile
    __shared__ float part[8][16][64];         // 32 KB: per-wave partial outputs

    int n0 = blockIdx.x * 16;
    int wave = threadIdx.x >> 5, lane = threadIdx.x & 31;
    int r = lane & 15, half = lane >> 4;

    float rmax = rowmax[n0 + r];              // A-fragment row for this lane

    uint32_t lds0 = (uint32_t)(uintptr_t)&astage[wave][0][0];
    uint32_t lds1 = (uint32_t)(uintptr_t)&astage[wave][1][0];

    // Stage one 16x32 f32 tile of S (rows n0..n0+15, cols kk..kk+31) into LDS.
    // 512 floats = 128 x 16B chunks; 32 lanes x 4 issues.
    auto stage = [&](uint32_t ldsbase, int kk) {
#pragma unroll
        for (int j = 0; j < 4; ++j) {
            int flat16 = j * 32 + lane;           // 16B chunk index
            int srow = flat16 >> 3;               // 8 chunks per 32-float row
            int c4   = (flat16 & 7) * 4;
            const float* g = S + (size_t)(n0 + srow) * NN + kk + c4;
            uint32_t ldso = ldsbase + (uint32_t)flat16 * 16u;
            asm volatile("global_load_async_to_lds_b128 %0, %1, off"
                         :: "v"(ldso), "v"(g) : "memory");
        }
    };

    v8f c[4] = {{}, {}, {}, {}};
    stage(lds0, wave * 32);
    for (int i = 0; i < 16; ++i) {
        int cur = i & 1;
        if (i + 1 < 16) {
            stage(cur ? lds0 : lds1, ((i + 1) * 8 + wave) * 32);
            asm volatile("s_wait_asynccnt 0x4" ::: "memory");  // prev stage done
        } else {
            asm volatile("s_wait_asynccnt 0x0" ::: "memory");
        }
        const float* ap = &astage[wave][cur][0] + r * 32;
        v16h a;
#pragma unroll
        for (int j = 0; j < 8; ++j) {
            a[j]     = (_Float16)__expf(ap[half * 8 + j]      - rmax);
            a[8 + j] = (_Float16)__expf(ap[16 + half * 8 + j] - rmax);
        }
        int kk = (i * 8 + wave) * 32;
#pragma unroll
        for (int ct = 0; ct < 4; ++ct) {
            v16h b = load_B16(V, HH, kk, hoff + ct * 16, lane);
            c[ct] = wmma32(a, b, c[ct]);
        }
    }

    int rloc = half * 8, cl = r;
#pragma unroll
    for (int ct = 0; ct < 4; ++ct)
#pragma unroll
        for (int rr = 0; rr < 8; ++rr)
            part[wave][rloc + rr][ct * 16 + cl] = c[ct][rr];
    __syncthreads();

#pragma unroll
    for (int e = 0; e < 4; ++e) {
        int flat = threadIdx.x + 256 * e;
        int row = flat >> 6, col = flat & 63;
        float s = 0.f;
#pragma unroll
        for (int w = 0; w < 8; ++w) s += part[w][row][col];
        out[(size_t)(n0 + row) * HH + hoff + col] = s / rowsum[n0 + row];
    }
}

// Final pooling: g = mean_n x; g = gelu(LN(g@W1+b1)); out = g@W2+b2
__global__ __launch_bounds__(256) void pool_kernel(
    const float* __restrict__ x,
    const float* __restrict__ W1, const float* __restrict__ b1,
    const float* __restrict__ g,  const float* __restrict__ beta,
    const float* __restrict__ W2, const float* __restrict__ b2,
    float* __restrict__ out)
{
    __shared__ float gm[256], t2[256], red[256];
    int t = threadIdx.x;

    float s = 0.f;
    for (int i = 0; i < NN; ++i) s += x[(size_t)i * HH + t];
    gm[t] = s * (1.0f / (float)NN);
    __syncthreads();

    float acc = b1[t];
    for (int k = 0; k < HH; ++k) acc += gm[k] * W1[(size_t)k * HH + t];

    red[t] = acc; __syncthreads();
    for (int st = 128; st > 0; st >>= 1) { if (t < st) red[t] += red[t + st]; __syncthreads(); }
    float mean = red[0] * (1.0f / 256.0f); __syncthreads();
    float d = acc - mean;
    red[t] = d * d; __syncthreads();
    for (int st = 128; st > 0; st >>= 1) { if (t < st) red[t] += red[t + st]; __syncthreads(); }
    float var = red[0] * (1.0f / 256.0f); __syncthreads();

    float v = d * rsqrtf(var + 1e-5f) * g[t] + beta[t];
    t2[t] = gelu_exact(v);
    __syncthreads();

    for (int o = t; o < 512; o += 256) {
        float a2 = b2[o];
        for (int k = 0; k < HH; ++k) a2 += t2[k] * W2[(size_t)k * 512 + o];
        out[o] = a2;
    }
}

// ---------------------------------------------------------------------------
extern "C" void kernel_launch(void* const* d_in, const int* in_sizes, int n_in,
                              void* d_out, int out_size, void* d_ws, size_t ws_size,
                              hipStream_t stream) {
    const float* node_features = (const float*)d_in[0];
    const float* adj           = (const float*)d_in[1];
    const float* edge_features = (const float*)d_in[2];
    const int*   edge_src      = (const int*)d_in[3];
    const int*   edge_tgt      = (const int*)d_in[4];
    const float* enc_W1   = (const float*)d_in[5];
    const float* enc_b1   = (const float*)d_in[6];
    const float* enc_g1   = (const float*)d_in[7];
    const float* enc_be1  = (const float*)d_in[8];
    const float* enc_W2   = (const float*)d_in[9];
    const float* enc_b2   = (const float*)d_in[10];
    const float* Wq = (const float*)d_in[11];
    const float* bq = (const float*)d_in[12];
    const float* Wk = (const float*)d_in[13];
    const float* bk = (const float*)d_in[14];
    const float* Wv = (const float*)d_in[15];
    const float* bv = (const float*)d_in[16];
    const float* We = (const float*)d_in[17];
    const float* be = (const float*)d_in[18];
    const float* Wo = (const float*)d_in[19];
    const float* bo = (const float*)d_in[20];
    const float* ln_g = (const float*)d_in[21];
    const float* ln_b = (const float*)d_in[22];
    const float* pool_W1 = (const float*)d_in[23];
    const float* pool_b1 = (const float*)d_in[24];
    const float* pool_g  = (const float*)d_in[25];
    const float* pool_be = (const float*)d_in[26];
    const float* pool_W2 = (const float*)d_in[27];
    const float* pool_b2 = (const float*)d_in[28];
    float* out = (float*)d_out;

    // workspace layout (floats): x | hbuf | Q | K | V | edge_attn | S | rowmax | rowsum
    float* ws  = (float*)d_ws;
    const size_t NH = (size_t)NN * HH;     // 1,048,576
    float* x    = ws;
    float* hbuf = ws + NH;                 // encoder mid / attention output
    float* Qb   = ws + 2 * NH;
    float* Kb   = ws + 3 * NH;
    float* Vb   = ws + 4 * NH;
    float* ea   = ws + 5 * NH;             // E*4
    float* S    = ws + 5 * NH + (size_t)EE * 4;   // NN*NN
    float* rmx  = S + (size_t)NN * NN;     // NN
    float* rsm  = rmx + NN;                // NN

    const int GEMM_GRID = NN / 16;         // 256 row-strips

    // Encoder: h = gelu(LN(nf@W1+b1)); x = h@W2+b2
    gemm16<1><<<GEMM_GRID, 256, 0, stream>>>(node_features, INDIM, INDIM,
        enc_W1, enc_b1, enc_g1, enc_be1, nullptr, hbuf);
    gemm16<0><<<GEMM_GRID, 256, 0, stream>>>(hbuf, HH, HH,
        enc_W2, enc_b2, nullptr, nullptr, nullptr, x);

    for (int l = 0; l < 3; ++l) {
        const size_t wl = (size_t)l * HH * HH;
        gemm16<0><<<GEMM_GRID, 256, 0, stream>>>(x, HH, HH, Wq + wl, bq + l * HH,
            nullptr, nullptr, nullptr, Qb);
        gemm16<0><<<GEMM_GRID, 256, 0, stream>>>(x, HH, HH, Wk + wl, bk + l * HH,
            nullptr, nullptr, nullptr, Kb);
        gemm16<0><<<GEMM_GRID, 256, 0, stream>>>(x, HH, HH, Wv + wl, bv + l * HH,
            nullptr, nullptr, nullptr, Vb);
        edge_attn_kernel<<<EE / 256, 256, 0, stream>>>(edge_features,
            We + (size_t)l * 64 * 4, be + l * 4, ea);

        for (int h = 0; h < 4; ++h) {
            int hoff = h * 64;
            qk_kernel<<<(NN / 16) * 8, 256, 0, stream>>>(Qb, Kb, adj, S, hoff);
            scatter_kernel<<<EE / 256, 256, 0, stream>>>(S, edge_src, edge_tgt, ea, h);
            rowstats_kernel<<<NN, 256, 0, stream>>>(S, rmx, rsm);
            av_kernel<<<NN / 16, 256, 0, stream>>>(S, Vb, rmx, rsm, hbuf, hoff);
        }

        gemm16<2><<<GEMM_GRID, 256, 0, stream>>>(hbuf, HH, HH, Wo + wl, bo + l * HH,
            ln_g + l * HH, ln_b + l * HH, x, x);
    }

    pool_kernel<<<1, 256, 0, stream>>>(x, pool_W1, pool_b1, pool_g, pool_be,
                                       pool_W2, pool_b2, out);
}
